// XposMultiHeadedAttention_27582279974925
// MI455X (gfx1250) — matrix operand order repaired and checked by reading, not compiled, not run on hardware
//
#include <hip/hip_runtime.h>
#include <hip/hip_bf16.h>

// ---------------- types ----------------
typedef __attribute__((ext_vector_type(16))) _Float16 v16h;
typedef __attribute__((ext_vector_type(8)))  _Float16 v8h;
typedef __attribute__((ext_vector_type(4)))  _Float16 v4h;
typedef __attribute__((ext_vector_type(8)))  float    v8f;

union F16Frag { v16h v; v8h h[2]; };

#define WMMA_F16(A_, B_, C_) \
  __builtin_amdgcn_wmma_f32_16x16x32_f16(false, (A_), false, (B_), (short)0, (C_), false, false)

// Async copy: 16B global -> LDS, tracked by ASYNCcnt (CDNA5 GLOBAL_LOAD_ASYNC_TO_LDS_B128).
// GVS addressing: SGPR64 base + per-lane 32-bit byte offset. LDS dest = low 32 bits of
// the flat shared pointer (ISA 10.2: LDS aperture truncates to addr[31:0]).
__device__ __forceinline__ void async_b128_to_lds(void* lds_dst, const void* gbase,
                                                  unsigned byte_off) {
  unsigned ldsa = (unsigned)(size_t)lds_dst;
  asm volatile("global_load_async_to_lds_b128 %0, %1, %2"
               :: "v"(ldsa), "v"(byte_off), "s"(gbase) : "memory");
}
__device__ __forceinline__ void wait_async0() {
  asm volatile("s_wait_asynccnt 0x0" ::: "memory");
}

// Problem constants (fixed by the reference)
static constexpr int TT   = 2048;   // sequence length
static constexpr int CC   = 1024;   // embed
static constexpr int HH   = 16;     // heads
static constexpr int HDIM = 64;     // head dim

// ============================================================================
// Kernel 1: projection GEMM  dst_f16[B][H][T][HD] = A_f32[4096x1024] @ W + bias
// WG tile 128x128, 8 waves (2x4), wave tile 64x32, K-step 32 (one WMMA depth)
// ============================================================================
__global__ __launch_bounds__(256) void qkv_gemm_kernel(
    const float* __restrict__ A, const float* __restrict__ W,
    const float* __restrict__ bias, _Float16* __restrict__ dst)
{
  constexpr int LDA = 40;  // padded halves per LDS row (80B: 16B-aligned, conflict-free)
  __shared__ alignas(16) _Float16 ldsA[128 * LDA];   // [m][k]
  __shared__ alignas(16) _Float16 ldsB[128 * LDA];   // [n][k] (transposed weight tile)

  const int tid  = threadIdx.x;
  const int wave = tid >> 5, lane = tid & 31;
  const int g    = lane >> 4, ln = lane & 15;
  const int wrow = wave >> 2;       // 0..1 -> m offset wrow*64
  const int wcol = wave & 3;        // 0..3 -> n offset wcol*32
  const int m0   = blockIdx.y * 128;
  const int n0   = blockIdx.x * 128;

  v8f acc[4][2];
#pragma unroll
  for (int i = 0; i < 4; ++i)
#pragma unroll
    for (int j = 0; j < 2; ++j) acc[i][j] = {};

  for (int k0 = 0; k0 < CC; k0 += 32) {
    // stage A tile 128x32 (f32 -> f16), coalesced float4 along k
#pragma unroll
    for (int p = 0; p < 4; ++p) {
      int idx = tid + p * 256;           // 1024 float4 chunks
      int r = idx >> 3, c4 = idx & 7;
      float4 f = *reinterpret_cast<const float4*>(A + (size_t)(m0 + r) * CC + k0 + c4 * 4);
      v4h h; h.x = (_Float16)f.x; h.y = (_Float16)f.y; h.z = (_Float16)f.z; h.w = (_Float16)f.w;
      *reinterpret_cast<v4h*>(&ldsA[r * LDA + c4 * 4]) = h;
    }
    // stage W tile 32x128 transposed into [n][k], coalesced float4 along n
#pragma unroll
    for (int p = 0; p < 4; ++p) {
      int idx = tid + p * 256;
      int k = idx >> 5, c4 = idx & 31;
      float4 f = *reinterpret_cast<const float4*>(W + (size_t)(k0 + k) * CC + n0 + c4 * 4);
      ldsB[(c4 * 4 + 0) * LDA + k] = (_Float16)f.x;
      ldsB[(c4 * 4 + 1) * LDA + k] = (_Float16)f.y;
      ldsB[(c4 * 4 + 2) * LDA + k] = (_Float16)f.z;
      ldsB[(c4 * 4 + 3) * LDA + k] = (_Float16)f.w;
    }
    __syncthreads();

    F16Frag af[4], bf[2];
#pragma unroll
    for (int i = 0; i < 4; ++i) {       // A frag: m=ln, elems 0-7 -> k=g*8.., 8-15 -> k=16+g*8..
      const _Float16* p = &ldsA[(wrow * 64 + i * 16 + ln) * LDA];
      af[i].h[0] = *reinterpret_cast<const v8h*>(p + g * 8);
      af[i].h[1] = *reinterpret_cast<const v8h*>(p + 16 + g * 8);
    }
#pragma unroll
    for (int j = 0; j < 2; ++j) {       // B frag: n=ln, elems 0-7 -> k=g*8.., 8-15 -> k=16+g*8..
      const _Float16* p = &ldsB[(wcol * 32 + j * 16 + ln) * LDA + g * 8];
      bf[j].h[0] = *reinterpret_cast<const v8h*>(p);
      bf[j].h[1] = *reinterpret_cast<const v8h*>(p + 16);
    }
#pragma unroll
    for (int i = 0; i < 4; ++i)
#pragma unroll
      for (int j = 0; j < 2; ++j)
        acc[i][j] = WMMA_F16(af[i].v, bf[j].v, acc[i][j]);
    __syncthreads();
  }

  // epilogue: +bias, scatter f16 to [B][H][T][HD]
#pragma unroll
  for (int i = 0; i < 4; ++i) {
    int mbase = m0 + wrow * 64 + i * 16;
#pragma unroll
    for (int j = 0; j < 2; ++j) {
      int col = n0 + wcol * 32 + j * 16 + ln;   // n = lane&15 in C layout
      int h = col >> 6, d = col & 63;
      float bv = bias[col];
#pragma unroll
      for (int v = 0; v < 8; ++v) {             // m = v + 8*(lane>>4)
        int row = mbase + v + 8 * g;
        int b = row >> 11, t = row & (TT - 1);
        dst[(((size_t)(b * HH + h)) * TT + t) * HDIM + d] = (_Float16)(acc[i][j][v] + bv);
      }
    }
  }
}

// ============================================================================
// Kernel 2: XPos rotary for Q (upscale, folded HD^-0.5) and K (downscale)
// one thread per (bh, t, pair i); min_pos = -T//2 = -1024, SCALE_BASE = 512
// ============================================================================
__global__ __launch_bounds__(256) void xpos_kernel(
    const _Float16* __restrict__ qraw, const _Float16* __restrict__ kraw,
    _Float16* __restrict__ qrot, _Float16* __restrict__ krot)
{
  int idx = blockIdx.x * blockDim.x + threadIdx.x;   // < 32*2048*32
  int i  = idx & 31;
  int t  = (idx >> 5) & (TT - 1);
  int bh = idx >> 16;

  float sv    = (2.0f * i + 0.4f * 64.0f) / (1.4f * 64.0f);
  float power = ((float)t - 1024.0f) * (1.0f / 512.0f);
  float scale = __expf(power * __logf(sv));                 // sv^power
  float invf  = __expf(-((float)i / 32.0f) * 9.2103403720f); // 10000^(-i/32)
  float ang   = (float)t * invf;
  float s = sinf(ang), c = cosf(ang);

  size_t base = ((size_t)bh * TT + t) * HDIM + 2 * i;
  float q0 = (float)qraw[base], q1 = (float)qraw[base + 1];
  float cs = c * scale, ss = s * scale;
  qrot[base]     = (_Float16)(0.125f * (q0 * cs - q1 * ss));   // 0.125 = HD^-0.5
  qrot[base + 1] = (_Float16)(0.125f * (q1 * cs + q0 * ss));

  float k0 = (float)kraw[base], k1 = (float)kraw[base + 1];
  float ci = c / scale, si = s / scale;                        // downscale
  krot[base]     = (_Float16)(k0 * ci - k1 * si);
  krot[base + 1] = (_Float16)(k1 * ci + k0 * si);
}

// ============================================================================
// Kernel 3: flash attention. grid (T/128, B*H), 256 threads = 8 waves.
// Each wave: 16 queries x HD=64 output, online softmax over 64-key blocks.
// K tile staged by async global->LDS DMA; V needs a transpose so it goes
// through VGPRs.
// ============================================================================
__global__ __launch_bounds__(256) void flash_attn_kernel(
    const _Float16* __restrict__ Q, const _Float16* __restrict__ K,
    const _Float16* __restrict__ V, const unsigned char* __restrict__ mask,
    _Float16* __restrict__ out)
{
  constexpr int LDK = 72;  // padded halves per row (144B: 16B-aligned, conflict-free)
  __shared__ alignas(16) _Float16 Klds[64 * LDK];        // [key][d]
  __shared__ alignas(16) _Float16 Vtlds[64 * LDK];       // [d][key]
  __shared__ alignas(16) _Float16 Plds[8 * 16 * LDK];    // per-wave P tile [m][key]

  const int bh = blockIdx.y;
  const int b  = bh >> 4, h = bh & 15;
  const int q0 = blockIdx.x * 128;
  const int tid = threadIdx.x, wave = tid >> 5, lane = tid & 31;
  const int g = lane >> 4, ln = lane & 15;
  const size_t headbase = (size_t)bh * TT * HDIM;
  const _Float16* gK = K + headbase;
  const _Float16* gV = V + headbase;

  // preload this wave's Q fragments (16 rows x 64 d) straight from global
  const _Float16* qp = Q + headbase + (size_t)(q0 + wave * 16 + ln) * HDIM;
  F16Frag qf[2];
  qf[0].h[0] = *reinterpret_cast<const v8h*>(qp + g * 8);
  qf[0].h[1] = *reinterpret_cast<const v8h*>(qp + 16 + g * 8);
  qf[1].h[0] = *reinterpret_cast<const v8h*>(qp + 32 + g * 8);
  qf[1].h[1] = *reinterpret_cast<const v8h*>(qp + 48 + g * 8);

  v8f o[4];
#pragma unroll
  for (int dt = 0; dt < 4; ++dt) o[dt] = {};
  float mrow[8], lrow[8];
#pragma unroll
  for (int v = 0; v < 8; ++v) { mrow[v] = -1e30f; lrow[v] = 0.0f; }

  _Float16* myP = Plds + wave * 16 * LDK;

  for (int kb = 0; kb < TT; kb += 64) {
    // 1) kick off async DMA of the K tile (pure copy, no VGPR round-trip)
#pragma unroll
    for (int p = 0; p < 2; ++p) {
      int idx = tid + p * 256;                  // 512 chunks of 8 halves
      int key = idx >> 3, c = (idx & 7) * 8;
      async_b128_to_lds(&Klds[key * LDK + c], gK,
                        (unsigned)(((kb + key) * HDIM + c) * 2));
    }
    // 2) meanwhile stage V transposed [d][key] through VGPRs
#pragma unroll
    for (int p = 0; p < 2; ++p) {
      int idx = tid + p * 256;
      int key = idx >> 3, c = (idx & 7) * 8;
      v8h vv = *reinterpret_cast<const v8h*>(gV + (size_t)(kb + key) * HDIM + c);
#pragma unroll
      for (int j = 0; j < 8; ++j) Vtlds[(c + j) * LDK + key] = vv[j];
    }
    wait_async0();
    __syncthreads();

    // S = Q K^T : 4 key-subtiles x 2 d-chunks
    v8f s[4];
#pragma unroll
    for (int kt = 0; kt < 4; ++kt) {
      v8f a = {};
      const _Float16* kp = &Klds[(kt * 16 + ln) * LDK];
      F16Frag k0f, k1f;
      k0f.h[0] = *reinterpret_cast<const v8h*>(kp + g * 8);
      k0f.h[1] = *reinterpret_cast<const v8h*>(kp + 16 + g * 8);
      k1f.h[0] = *reinterpret_cast<const v8h*>(kp + 32 + g * 8);
      k1f.h[1] = *reinterpret_cast<const v8h*>(kp + 48 + g * 8);
      a = WMMA_F16(qf[0].v, k0f.v, a);
      a = WMMA_F16(qf[1].v, k1f.v, a);
      s[kt] = a;
    }

    // key padding mask: column n = lane&15 within each subtile
    const unsigned char* mp = mask + (size_t)b * TT + kb;
#pragma unroll
    for (int kt = 0; kt < 4; ++kt)
      if (mp[kt * 16 + ln]) {
#pragma unroll
        for (int v = 0; v < 8; ++v) s[kt][v] = -1e30f;
      }

    // online softmax (rows replicated across the 16-lane half)
#pragma unroll
    for (int v = 0; v < 8; ++v) {
      float mx = fmaxf(fmaxf(s[0][v], s[1][v]), fmaxf(s[2][v], s[3][v]));
#pragma unroll
      for (int off = 1; off < 16; off <<= 1) mx = fmaxf(mx, __shfl_xor(mx, off, 32));
      float mn = fmaxf(mrow[v], mx);
      float corr = __expf(mrow[v] - mn);
      mrow[v] = mn;
      lrow[v] *= corr;
#pragma unroll
      for (int dt = 0; dt < 4; ++dt) o[dt][v] *= corr;
      float rs = 0.0f;
#pragma unroll
      for (int kt = 0; kt < 4; ++kt) {
        float pe = __expf(s[kt][v] - mn);
        s[kt][v] = pe;
        rs += pe;
      }
#pragma unroll
      for (int off = 1; off < 16; off <<= 1) rs += __shfl_xor(rs, off, 32);
      lrow[v] += rs;
    }

    // re-layout P (C layout -> A fragments) via wave-private LDS
#pragma unroll
    for (int kt = 0; kt < 4; ++kt)
#pragma unroll
      for (int v = 0; v < 8; ++v)
        myP[(v + 8 * g) * LDK + kt * 16 + ln] = (_Float16)s[kt][v];

    // O += P V : 2 key-chunks x 4 d-tiles
#pragma unroll
    for (int kc = 0; kc < 2; ++kc) {
      F16Frag pf;
      const _Float16* pp = myP + ln * LDK + kc * 32;
      pf.h[0] = *reinterpret_cast<const v8h*>(pp + g * 8);
      pf.h[1] = *reinterpret_cast<const v8h*>(pp + 16 + g * 8);
#pragma unroll
      for (int dt = 0; dt < 4; ++dt) {
        F16Frag vf;
        const _Float16* vp = &Vtlds[(dt * 16 + ln) * LDK + kc * 32 + g * 8];
        vf.h[0] = *reinterpret_cast<const v8h*>(vp);
        vf.h[1] = *reinterpret_cast<const v8h*>(vp + 16);
        o[dt] = WMMA_F16(pf.v, vf.v, o[dt]);
      }
    }
    __syncthreads();
  }

  // normalize and store as f16 into [B*T][C] (col = h*64 + d)
#pragma unroll
  for (int dt = 0; dt < 4; ++dt)
#pragma unroll
    for (int v = 0; v < 8; ++v) {
      int t = q0 + wave * 16 + v + 8 * g;
      int d = dt * 16 + ln;
      out[((size_t)(b * TT + t)) * CC + h * HDIM + d] = (_Float16)(o[dt][v] / lrow[v]);
    }
}

// ============================================================================
// Kernel 4: output projection  out_f32[4096x1024] = attn_f16 @ Wo(f32->f16) + bo
// A tile is a pure f16 copy -> async global->LDS DMA; Wo converts via VALU.
// ============================================================================
__global__ __launch_bounds__(256) void out_gemm_kernel(
    const _Float16* __restrict__ A, const float* __restrict__ W,
    const float* __restrict__ bias, float* __restrict__ dst)
{
  constexpr int LDA = 40;
  __shared__ alignas(16) _Float16 ldsA[128 * LDA];
  __shared__ alignas(16) _Float16 ldsB[128 * LDA];

  const int tid  = threadIdx.x;
  const int wave = tid >> 5, lane = tid & 31;
  const int g    = lane >> 4, ln = lane & 15;
  const int wrow = wave >> 2, wcol = wave & 3;
  const int m0   = blockIdx.y * 128;
  const int n0   = blockIdx.x * 128;

  v8f acc[4][2];
#pragma unroll
  for (int i = 0; i < 4; ++i)
#pragma unroll
    for (int j = 0; j < 2; ++j) acc[i][j] = {};

  for (int k0 = 0; k0 < CC; k0 += 32) {
    // A already f16: 512 x 16B chunks via async DMA
#pragma unroll
    for (int p = 0; p < 2; ++p) {
      int idx = tid + p * 256;
      int r = idx >> 2, c8 = (idx & 3) * 8;
      async_b128_to_lds(&ldsA[r * LDA + c8], A,
                        (unsigned)((((size_t)(m0 + r)) * CC + k0 + c8) * 2));
    }
#pragma unroll
    for (int p = 0; p < 4; ++p) {                  // Wo f32 -> f16, transposed
      int idx = tid + p * 256;
      int k = idx >> 5, c4 = idx & 31;
      float4 f = *reinterpret_cast<const float4*>(W + (size_t)(k0 + k) * CC + n0 + c4 * 4);
      ldsB[(c4 * 4 + 0) * LDA + k] = (_Float16)f.x;
      ldsB[(c4 * 4 + 1) * LDA + k] = (_Float16)f.y;
      ldsB[(c4 * 4 + 2) * LDA + k] = (_Float16)f.z;
      ldsB[(c4 * 4 + 3) * LDA + k] = (_Float16)f.w;
    }
    wait_async0();
    __syncthreads();

    F16Frag af[4], bf[2];
#pragma unroll
    for (int i = 0; i < 4; ++i) {
      const _Float16* p = &ldsA[(wrow * 64 + i * 16 + ln) * LDA];
      af[i].h[0] = *reinterpret_cast<const v8h*>(p + g * 8);
      af[i].h[1] = *reinterpret_cast<const v8h*>(p + 16 + g * 8);
    }
#pragma unroll
    for (int j = 0; j < 2; ++j) {
      const _Float16* p = &ldsB[(wcol * 32 + j * 16 + ln) * LDA + g * 8];
      bf[j].h[0] = *reinterpret_cast<const v8h*>(p);
      bf[j].h[1] = *reinterpret_cast<const v8h*>(p + 16);
    }
#pragma unroll
    for (int i = 0; i < 4; ++i)
#pragma unroll
      for (int j = 0; j < 2; ++j)
        acc[i][j] = WMMA_F16(af[i].v, bf[j].v, acc[i][j]);
    __syncthreads();
  }

#pragma unroll
  for (int i = 0; i < 4; ++i) {
    int mbase = m0 + wrow * 64 + i * 16;
#pragma unroll
    for (int j = 0; j < 2; ++j) {
      int col = n0 + wcol * 32 + j * 16 + ln;
      float bv = bias[col];
#pragma unroll
      for (int v = 0; v < 8; ++v) {
        int row = mbase + v + 8 * g;
        dst[(size_t)row * CC + col] = acc[i][j][v] + bv;
      }
    }
  }
}

// ============================================================================
// launch
// ============================================================================
extern "C" void kernel_launch(void* const* d_in, const int* in_sizes, int n_in,
                              void* d_out, int out_size, void* d_ws, size_t ws_size,
                              hipStream_t stream) {
  (void)in_sizes; (void)n_in; (void)out_size; (void)ws_size;
  const float* query = (const float*)d_in[0];
  const float* key   = (const float*)d_in[1];
  const float* value = (const float*)d_in[2];
  const unsigned char* mask = (const unsigned char*)d_in[3];
  const float* Wq = (const float*)d_in[4];
  const float* bq = (const float*)d_in[5];
  const float* Wk = (const float*)d_in[6];
  const float* bk = (const float*)d_in[7];
  const float* Wv = (const float*)d_in[8];
  const float* bv = (const float*)d_in[9];
  const float* Wo = (const float*)d_in[10];
  const float* bo = (const float*)d_in[11];
  float* out = (float*)d_out;

  char* ws = (char*)d_ws;
  const size_t SZ = (size_t)4096 * 1024 * 2;   // 8 MB per f16 tensor
  _Float16* Qraw = (_Float16*)(ws);
  _Float16* Kraw = (_Float16*)(ws + SZ);
  _Float16* Vbuf = (_Float16*)(ws + 2 * SZ);
  _Float16* Qrot = (_Float16*)(ws + 3 * SZ);
  _Float16* Krot = (_Float16*)(ws + 4 * SZ);
  _Float16* Attn = (_Float16*)(ws + 5 * SZ);

  dim3 blk(256);
  dim3 gProj(8, 32);   // N/128, M/128

  qkv_gemm_kernel<<<gProj, blk, 0, stream>>>(query, Wq, bq, Qraw);
  qkv_gemm_kernel<<<gProj, blk, 0, stream>>>(key,   Wk, bk, Kraw);
  qkv_gemm_kernel<<<gProj, blk, 0, stream>>>(value, Wv, bv, Vbuf);

  xpos_kernel<<<dim3(8192), blk, 0, stream>>>(Qraw, Kraw, Qrot, Krot);

  flash_attn_kernel<<<dim3(16, 32), blk, 0, stream>>>(Qrot, Krot, Vbuf, mask, Attn);

  out_gemm_kernel<<<gProj, blk, 0, stream>>>(Attn, Wo, bo, out);
}
